// ArcFace_68942815035858
// MI455X (gfx1250) — matrix operand order repaired and checked by reading, not compiled, not run on hardware
//
#include <hip/hip_runtime.h>
#include <hip/hip_bf16.h>

typedef _Float16 v16h __attribute__((ext_vector_type(16)));
typedef _Float16 v8h  __attribute__((ext_vector_type(8)));
typedef float    v8f  __attribute__((ext_vector_type(8)));
typedef unsigned int uv4 __attribute__((ext_vector_type(4)));
typedef int          iv8 __attribute__((ext_vector_type(8)));
typedef int          iv4 __attribute__((ext_vector_type(4)));

#define B_ROWS 512
#define D_DIM  512
#define C_CLS  100000
#define N_TILES (C_CLS / 16)      // 6250 exact
#define K_STEPS (D_DIM / 32)      // 16
#define MB      128               // M rows per block in GEMM
#define LDSROW  40                // f16 per LDS row (32 data + 8 pad -> conflict-free b128)
#define LOGIT_S 64.0f

#if defined(__gfx1250__) && __has_builtin(__builtin_amdgcn_tensor_load_to_lds)
#define HAVE_TDM 1
#else
#define HAVE_TDM 0
#endif

union Frag16 { v16h v; v8h h[2]; };

__device__ __forceinline__ float wave_allsum32(float v) {
    #pragma unroll
    for (int m = 16; m >= 1; m >>= 1) v += __shfl_xor(v, m, 32);
    return v;
}

#if HAVE_TDM
// Issue a TDM load of a [128 rows x 32 halves] tile of xn into LDS.
// Hardware padding: after every 16 DWORDs (one 32-half row), insert 4 DWORDs
// (16B) -> LDS row stride 40 halves, matching the conflict-free layout.
__device__ __forceinline__ void tdm_stage_tile(const _Float16* gsrc,
                                               unsigned lds_byte_addr) {
    unsigned long long ga = (unsigned long long)(uintptr_t)gsrc;
    uv4 g0; iv8 g1; iv4 g2, g3; iv8 g4;
    g0[0] = 1u;                                   // count=1, user descriptor
    g0[1] = lds_byte_addr;                        // LDS dest (bytes)
    g0[2] = (unsigned)ga;                         // global_addr[31:0]
    g0[3] = ((unsigned)(ga >> 32) & 0x01FFFFFFu)  // global_addr[56:32]
          | (2u << 30);                           // type = 2 ("image")
    g1[0] = (1 << 16)                             // data_size = 2 bytes
          | (1 << 20)                             // pad_enable
          | (3 << 22)                             // pad_interval: 16 DWORDs
          | (3 << 25);                            // pad_amount: 4 DWORDs
    g1[1] = (int)((unsigned)(D_DIM & 0xFFFF) << 16);   // tensor_dim0 = 512 (lo16)
    g1[2] = (int)((unsigned)(D_DIM & 0xFFFF) << 16);   // dim0 hi16=0 | tensor_dim1 lo16 = 512
    g1[3] = (int)(32u << 16);                     // tensor_dim1 hi16=0 | tile_dim0 = 32
    g1[4] = (int)(unsigned)MB;                    // tile_dim1 = 128 | tile_dim2 = 0
    g1[5] = (int)(unsigned)D_DIM;                 // tensor_dim0_stride = 512
    g1[6] = 0;                                    // stride hi | tensor_dim1_stride lo
    g1[7] = 0;
    g2[0] = 0; g2[1] = 0; g2[2] = 0; g2[3] = 0;   // dims 2+ unused (2-D tile)
    g3[0] = 0; g3[1] = 0; g3[2] = 0; g3[3] = 0;
    g4[0] = 0; g4[1] = 0; g4[2] = 0; g4[3] = 0;   // extra group (6-arg form): unused
    g4[4] = 0; g4[5] = 0; g4[6] = 0; g4[7] = 0;
    __builtin_amdgcn_tensor_load_to_lds(g0, g1, g2, g3, g4, 0);
}
#endif

// ---------------------------------------------------------------------------
// Row-normalize fp32 [nrows, 512] -> f16, one wave per row.
// Optionally zero a per-row accumulator array (used for x pass).
// ---------------------------------------------------------------------------
__global__ void af_normalize_rows(const float* __restrict__ src,
                                  _Float16* __restrict__ dst,
                                  float* __restrict__ row_sum_zero,
                                  int nrows) {
    int wave = threadIdx.x >> 5;
    int lane = threadIdx.x & 31;
    int row  = blockIdx.x * 8 + wave;
    if (row >= nrows) return;

    const float4* p4 = (const float4*)(src + (size_t)row * D_DIM + lane * 16);
    float buf[16];
    #pragma unroll
    for (int q = 0; q < 4; ++q) {
        float4 f = p4[q];
        buf[q*4+0] = f.x; buf[q*4+1] = f.y; buf[q*4+2] = f.z; buf[q*4+3] = f.w;
    }
    float ss = 0.f;
    #pragma unroll
    for (int j = 0; j < 16; ++j) ss += buf[j] * buf[j];
    ss = wave_allsum32(ss);
    float inv = 1.0f / fmaxf(sqrtf(ss), 1e-12f);

    v8h h0, h1;
    #pragma unroll
    for (int j = 0; j < 8; ++j) {
        h0[j] = (_Float16)(buf[j]     * inv);
        h1[j] = (_Float16)(buf[j + 8] * inv);
    }
    v8h* o = (v8h*)(dst + (size_t)row * D_DIM + lane * 16);
    o[0] = h0; o[1] = h1;

    if (row_sum_zero && lane == 0) row_sum_zero[row] = 0.0f;
}

// ---------------------------------------------------------------------------
// Label logits: out[b] = S * dot(xn[b], wn[label[b]]), one wave per row.
// ---------------------------------------------------------------------------
__global__ void af_label_logits(const _Float16* __restrict__ xn,
                                const _Float16* __restrict__ wn,
                                const int* __restrict__ label,
                                float* __restrict__ out) {
    int wave = threadIdx.x >> 5;
    int lane = threadIdx.x & 31;
    int b = blockIdx.x * 8 + wave;
    if (b >= B_ROWS) return;
    int c = label[b];

    const v8h* xp = (const v8h*)(xn + (size_t)b * D_DIM + lane * 16);
    const v8h* wp = (const v8h*)(wn + (size_t)c * D_DIM + lane * 16);
    v8h x0 = xp[0], x1 = xp[1], w0 = wp[0], w1 = wp[1];
    float s = 0.f;
    #pragma unroll
    for (int j = 0; j < 8; ++j) {
        s += (float)x0[j] * (float)w0[j];
        s += (float)x1[j] * (float)w1[j];
    }
    s = wave_allsum32(s);
    if (lane == 0) out[b] = LOGIT_S * s;
}

// ---------------------------------------------------------------------------
// WMMA GEMM + streamed sum-of-exp, TDM double-buffered A staging.
// Block: 256 threads (8 waves). blockIdx.x = M-block (128 rows),
// blockIdx.y*8 + wave = 16-class column tile. Each wave: 8 M-tiles x 1 N-tile,
// K=512 in 16 steps of v_wmma_f32_16x16x32_f16.
// Wave 0 issues tensor_load_to_lds for step k+1 while all waves compute step k.
// Epilogue accumulates exp(S*cos - 64) per row via f32 atomics.
// ---------------------------------------------------------------------------
__global__ void __launch_bounds__(256)
af_gemm_lse(const _Float16* __restrict__ xn,
            const _Float16* __restrict__ wn,
            float* __restrict__ row_sum) {
    __shared__ _Float16 ldsA[2][MB * LDSROW];

    const int tid  = threadIdx.x;
    const int wave = tid >> 5;
    const int lane = tid & 31;
    const int lh   = lane >> 4;      // K-half selector
    const int r    = lane & 15;      // row-in-tile / column-in-tile

    const int rowBase = blockIdx.x * MB;
    const int ntile   = blockIdx.y * 8 + wave;
    const bool valid  = ntile < N_TILES;     // wave-uniform -> EXEC stays full

    v8f acc[8];
    #pragma unroll
    for (int t = 0; t < 8; ++t)
        #pragma unroll
        for (int i = 0; i < 8; ++i) acc[t][i] = 0.0f;

    const int cls = ntile * 16 + r;  // class index for this lane's B column
    const _Float16* wbase = wn + (size_t)(valid ? cls : 0) * D_DIM;
    const _Float16* atile = xn + (size_t)rowBase * D_DIM;

#if HAVE_TDM
    const unsigned ldsBase0 = (unsigned)(uintptr_t)(void*)&ldsA[0][0];
    const unsigned ldsBase1 = (unsigned)(uintptr_t)(void*)&ldsA[1][0];

    if (wave == 0) {
        tdm_stage_tile(atile, ldsBase0);             // prologue: stage k-step 0
        __builtin_amdgcn_s_wait_tensorcnt(0);
    }
    __syncthreads();

    for (int ks = 0; ks < K_STEPS; ++ks) {
        const int k0  = ks * 32;
        const int cur = ks & 1;
        if (wave == 0 && ks + 1 < K_STEPS)           // async: stage next tile
            tdm_stage_tile(atile + (k0 + 32), cur ? ldsBase0 : ldsBase1);

        if (valid) {
            Frag16 bf;
            const v8h* gb = (const v8h*)(wbase + k0 + lh * 8);
            bf.h[0] = gb[0]; bf.h[1] = gb[2];
            if (ks + 1 < K_STEPS)
                __builtin_prefetch(wbase + k0 + 32 + lh * 16, 0, 1);

            const _Float16* lbuf = &ldsA[cur][0];
            #pragma unroll
            for (int t = 0; t < 8; ++t) {
                Frag16 af;
                const _Float16* lb = lbuf + (t * 16 + r) * LDSROW + lh * 8;
                af.h[0] = *(const v8h*)lb;
                af.h[1] = *(const v8h*)(lb + 16);
                acc[t] = __builtin_amdgcn_wmma_f32_16x16x32_f16(
                    false, af.v, false, bf.v, (short)0, acc[t], false, false);
            }
        }
        if (wave == 0)
            __builtin_amdgcn_s_wait_tensorcnt(0);    // next buffer fully landed
        __syncthreads();
    }
#else
    // Fallback: synchronous cooperative staging (single buffer).
    const int srow  = tid >> 1;
    const int shalf = tid & 1;
    for (int ks = 0; ks < K_STEPS; ++ks) {
        const int k0 = ks * 32;
        __syncthreads();
        {
            const v8h* g = (const v8h*)(atile + (size_t)srow * D_DIM + k0 + shalf * 16);
            v8h* l = (v8h*)(&ldsA[0][0] + srow * LDSROW + shalf * 16);
            l[0] = g[0]; l[1] = g[1];
        }
        __syncthreads();
        if (valid) {
            Frag16 bf;
            const v8h* gb = (const v8h*)(wbase + k0 + lh * 8);
            bf.h[0] = gb[0]; bf.h[1] = gb[2];
            #pragma unroll
            for (int t = 0; t < 8; ++t) {
                Frag16 af;
                const _Float16* lb = &ldsA[0][0] + (t * 16 + r) * LDSROW + lh * 8;
                af.h[0] = *(const v8h*)lb;
                af.h[1] = *(const v8h*)(lb + 16);
                acc[t] = __builtin_amdgcn_wmma_f32_16x16x32_f16(
                    false, af.v, false, bf.v, (short)0, acc[t], false, false);
            }
        }
    }
#endif

    if (valid) {
        #pragma unroll
        for (int t = 0; t < 8; ++t) {
            #pragma unroll
            for (int i = 0; i < 8; ++i) {
                // C layout: lanes 0-15 -> row i, lanes 16-31 -> row i+8
                float v = expf(LOGIT_S * acc[t][i] - 64.0f);
                #pragma unroll
                for (int m = 1; m < 16; m <<= 1) v += __shfl_xor(v, m, 32);
                if (r == 0) {
                    int row = rowBase + t * 16 + i + lh * 8;
                    atomicAdd(&row_sum[row], v);
                }
            }
        }
    }
}

// ---------------------------------------------------------------------------
// loss = mean_b( 64 + log(sum_b) - label_logit[b] )
// ---------------------------------------------------------------------------
__global__ void af_final_loss(const float* __restrict__ row_sum,
                              const float* __restrict__ label_logit,
                              float* __restrict__ out) {
    __shared__ float red[256];
    int tid = threadIdx.x;
    float s = 0.f;
    for (int b = tid; b < B_ROWS; b += 256)
        s += (64.0f + logf(row_sum[b])) - label_logit[b];
    red[tid] = s;
    __syncthreads();
    for (int off = 128; off > 0; off >>= 1) {
        if (tid < off) red[tid] += red[tid + off];
        __syncthreads();
    }
    if (tid == 0) out[0] = red[0] / (float)B_ROWS;
}

// ---------------------------------------------------------------------------
extern "C" void kernel_launch(void* const* d_in, const int* in_sizes, int n_in,
                              void* d_out, int out_size, void* d_ws, size_t ws_size,
                              hipStream_t stream) {
    const float* x     = (const float*)d_in[0];  // [512, 512]
    const float* w     = (const float*)d_in[1];  // [100000, 512]
    const int*   label = (const int*)d_in[2];    // [512]
    (void)in_sizes; (void)n_in; (void)out_size; (void)ws_size;

    char* ws = (char*)d_ws;
    _Float16* xn = (_Float16*)ws;                                   // 512*512 f16
    _Float16* wn = (_Float16*)(ws + (size_t)B_ROWS * D_DIM * 2);    // 100000*512 f16
    float* row_sum = (float*)(ws + (size_t)B_ROWS * D_DIM * 2
                                 + (size_t)C_CLS * D_DIM * 2);      // [512] f32
    float* lab_logit = row_sum + B_ROWS;                            // [512] f32

    af_normalize_rows<<<B_ROWS / 8, 256, 0, stream>>>(x, xn, row_sum, B_ROWS);
    af_normalize_rows<<<C_CLS / 8, 256, 0, stream>>>(w, wn, nullptr, C_CLS);
    af_label_logits<<<B_ROWS / 8, 256, 0, stream>>>(xn, wn, label, lab_logit);

    dim3 grid(B_ROWS / MB, (N_TILES + 7) / 8);   // 4 x 782
    af_gemm_lse<<<grid, 256, 0, stream>>>(xn, wn, row_sum);

    af_final_loss<<<1, 256, 0, stream>>>(row_sum, lab_logit, (float*)d_out);
}
